// Layer_3298534884152
// MI455X (gfx1250) — matrix operand order repaired and checked by reading, not compiled, not run on hardware
//
#include <hip/hip_runtime.h>

// ---------------------------------------------------------------------------
// W4A32 grouped-quant GEMM for MI455X (gfx1250, wave32, WMMA).
//
// C(4096x4096,f32) = A(4096x4096,f32) @ W, W = (nib4 - 8) * s[k/128, n]
// q packs 8 nibbles per int32 along N: nib(k, 8*w+j) = (q[k,w] >> 4j) & 0xF
//
// Compute-bound (137 GFLOP vs ~137 MB of traffic ~ 6us @ 23.3 TB/s), so we
// ride the bf16 WMMA pipe (16x16x32, f32 accum) with bf16x3 splitting for
// near-fp32 accuracy:  A@W ~= Ahi@Whi + Alo@Whi + Ahi@Wlo.
// ---------------------------------------------------------------------------

typedef __attribute__((ext_vector_type(16))) __bf16 v16bf;
typedef __attribute__((ext_vector_type(8)))  __bf16 v8bf;
typedef __attribute__((ext_vector_type(8)))  float  v8f;
typedef __attribute__((ext_vector_type(4)))  float  v4f;

#define M_DIM 4096
#define N_DIM 4096
#define K_DIM 4096
#define GROUP 128
#define NW    (N_DIM / 8)        // q words per K-row

#define TILE_M 128
#define TILE_N 128
#define TILE_K 32
#define LDSS   40                // padded K stride (elems): 20-word bank stride

__device__ __forceinline__ v16bf frag16(const __bf16* p0, const __bf16* p1) {
    v8bf x = *reinterpret_cast<const v8bf*>(p0);
    v8bf y = *reinterpret_cast<const v8bf*>(p1);
    return __builtin_shufflevector(x, y, 0,1,2,3,4,5,6,7,8,9,10,11,12,13,14,15);
}

__device__ __forceinline__ v8f wmma_bf16(v16bf a, v16bf b, v8f c) {
    return __builtin_amdgcn_wmma_f32_16x16x32_bf16(
        /*neg_a=*/false, a, /*neg_b=*/false, b,
        /*c_mod=*/(short)0, c, /*reuse_a=*/false, /*reuse_b=*/false);
}

__global__ __launch_bounds__(256, 2)
void w4a32_wmma_kernel(const float* __restrict__ A,
                       const float* __restrict__ S,
                       const int*   __restrict__ Q,
                       float*       __restrict__ C)
{
    __shared__ __align__(16) __bf16 lds_a_hi[TILE_M * LDSS];
    __shared__ __align__(16) __bf16 lds_a_lo[TILE_M * LDSS];
    __shared__ __align__(16) __bf16 lds_b_hi[TILE_N * LDSS];   // [n][k] layout
    __shared__ __align__(16) __bf16 lds_b_lo[TILE_N * LDSS];

    const int tid  = threadIdx.x;
    const int lane = tid & 31;
    const int wave = tid >> 5;
    const int wm   = wave >> 1;          // 0..3 : 32-row strip
    const int wn   = wave & 1;           // 0..1 : 64-col strip
    const int m0   = blockIdx.y * TILE_M;
    const int n0   = blockIdx.x * TILE_N;

    // ---- A staging assignment: one half K-row of the A tile per thread ----
    const int a_row   = tid >> 1;        // 0..127
    const int a_khalf = tid & 1;         // 0..1 (16 floats each)
    const float* a_src = A + (size_t)(m0 + a_row) * K_DIM + a_khalf * 16;
    __bf16* a_hi_dst = lds_a_hi + a_row * LDSS + a_khalf * 16;
    __bf16* a_lo_dst = lds_a_lo + a_row * LDSS + a_khalf * 16;

    // ---- fragment addressing (per ISA VGPR layouts, wave32) ----
    const int half     = (lane < 16) ? 0 : 1;
    const int lrow     = lane & 15;
    // A 16-bit 16x32: lane r -> K {0..7,16..23}; lane 16+r -> K {8..15,24..31}
    const int a_kbase  = half * 8;
    // B: lane c -> K {0..7,16..23} of col c; lane 16+c -> K {8..15,24..31}
    const int b_kbase  = half * 8;

    v8f acc[2][4];
#pragma unroll
    for (int mt = 0; mt < 2; ++mt)
#pragma unroll
        for (int nt = 0; nt < 4; ++nt) { v8f z = {}; acc[mt][nt] = z; }

    for (int g = 0; g < K_DIM / GROUP; ++g) {
        // scales for this thread's two dequant tasks (8 cols each), hoisted
        float sc[2][8], msc[2][8];
#pragma unroll
        for (int u = 0; u < 2; ++u) {
            const int id   = (tid << 1) | u;
            const int wcol = id & 15;
            const float* sp = S + (size_t)g * N_DIM + n0 + wcol * 8;
            v4f s0 = *reinterpret_cast<const v4f*>(sp);
            v4f s1 = *reinterpret_cast<const v4f*>(sp + 4);
#pragma unroll
            for (int j = 0; j < 4; ++j) {
                sc[u][j]     = s0[j];  msc[u][j]     = -8.0f * s0[j];
                sc[u][4 + j] = s1[j];  msc[u][4 + j] = -8.0f * s1[j];
            }
        }

        for (int kk4 = 0; kk4 < GROUP / TILE_K; ++kk4) {
            const int kk = g * GROUP + kk4 * TILE_K;

            // ---- stage A tile -> bf16 hi/lo in LDS (converted once) ----
            {
                const float* ap = a_src + kk;
                v8bf hi0, hi1, lo0, lo1;
#pragma unroll
                for (int q4 = 0; q4 < 4; ++q4) {
                    v4f f = *reinterpret_cast<const v4f*>(ap + q4 * 4);
#pragma unroll
                    for (int j = 0; j < 4; ++j) {
                        const int i  = q4 * 4 + j;
                        __bf16 h     = (__bf16)f[j];
                        __bf16 l     = (__bf16)(f[j] - (float)h);
                        if (i < 8) { hi0[i] = h; lo0[i] = l; }
                        else       { hi1[i - 8] = h; lo1[i - 8] = l; }
                    }
                }
                *reinterpret_cast<v8bf*>(a_hi_dst)     = hi0;
                *reinterpret_cast<v8bf*>(a_hi_dst + 8) = hi1;
                *reinterpret_cast<v8bf*>(a_lo_dst)     = lo0;
                *reinterpret_cast<v8bf*>(a_lo_dst + 8) = lo1;
            }

            // ---- dequant B tile -> bf16 hi/lo, [n][k] in LDS ----
#pragma unroll
            for (int u = 0; u < 2; ++u) {
                const int id   = (tid << 1) | u;
                const int krow = id >> 4;          // 0..31
                const int wcol = id & 15;          // word col in tile
                const unsigned qw =
                    (unsigned)Q[(size_t)(kk + krow) * NW + (n0 >> 3) + wcol];
#pragma unroll
                for (int j = 0; j < 8; ++j) {
                    const unsigned nib = (qw >> (4 * j)) & 0xFu;
                    const float w  = fmaf((float)nib, sc[u][j], msc[u][j]);
                    const __bf16 h = (__bf16)w;
                    const __bf16 l = (__bf16)(w - (float)h);
                    const int idx  = (wcol * 8 + j) * LDSS + krow;
                    lds_b_hi[idx] = h;
                    lds_b_lo[idx] = l;
                }
            }

            __syncthreads();

            // ---- fragments + WMMA: 24 x v_wmma_f32_16x16x32_bf16 / wave ----
            v16bf fa_hi[2], fa_lo[2];
#pragma unroll
            for (int mt = 0; mt < 2; ++mt) {
                const int r = (wm * 32 + mt * 16 + lrow) * LDSS + a_kbase;
                fa_hi[mt] = frag16(lds_a_hi + r, lds_a_hi + r + 16);
                fa_lo[mt] = frag16(lds_a_lo + r, lds_a_lo + r + 16);
            }
#pragma unroll
            for (int nt = 0; nt < 4; ++nt) {
                const int c = (wn * 64 + nt * 16 + lrow) * LDSS + b_kbase;
                v16bf fb_hi = frag16(lds_b_hi + c, lds_b_hi + c + 16);
                v16bf fb_lo = frag16(lds_b_lo + c, lds_b_lo + c + 16);
#pragma unroll
                for (int mt = 0; mt < 2; ++mt) {
                    acc[mt][nt] = wmma_bf16(fa_hi[mt], fb_hi, acc[mt][nt]);
                    acc[mt][nt] = wmma_bf16(fa_lo[mt], fb_hi, acc[mt][nt]);
                    acc[mt][nt] = wmma_bf16(fa_hi[mt], fb_lo, acc[mt][nt]);
                }
            }

            __syncthreads();   // before next iteration overwrites LDS
        }
    }

    // ---- writeback: C/D f32 layout: lanes 0-15 rows v, lanes 16-31 rows 8+v
#pragma unroll
    for (int mt = 0; mt < 2; ++mt) {
#pragma unroll
        for (int nt = 0; nt < 4; ++nt) {
            const int col = n0 + wn * 64 + nt * 16 + lrow;
            const int row = m0 + wm * 32 + mt * 16 + half * 8;
#pragma unroll
            for (int v = 0; v < 8; ++v) {
                C[(size_t)(row + v) * N_DIM + col] = acc[mt][nt][v];
            }
        }
    }
}

extern "C" void kernel_launch(void* const* d_in, const int* in_sizes, int n_in,
                              void* d_out, int out_size, void* d_ws, size_t ws_size,
                              hipStream_t stream) {
    (void)in_sizes; (void)n_in; (void)out_size; (void)d_ws; (void)ws_size;
    const float* A = (const float*)d_in[0];
    const float* S = (const float*)d_in[1];
    const int*   Q = (const int*)d_in[2];
    float*       C = (float*)d_out;

    dim3 grid(N_DIM / TILE_N, M_DIM / TILE_M);   // 32 x 32 blocks
    dim3 block(256);                              // 8 wave32 waves / block
    w4a32_wmma_kernel<<<grid, block, 0, stream>>>(A, S, Q, C);
}
